// TMSAB_7318624272592
// MI455X (gfx1250) — matrix operand order repaired and checked by reading, not compiled, not run on hardware
//
#include <hip/hip_runtime.h>
#include <hip/hip_bf16.h>

typedef __attribute__((ext_vector_type(16))) __bf16 v16bf;
typedef __attribute__((ext_vector_type(8)))  __bf16 v8bf;
typedef __attribute__((ext_vector_type(8)))  float  v8f;

#define DMODEL   512
#define NSEQ     2048
#define NHEADS   8
#define HDIM     64
#define NBATCH   2
#define NROWS    (NBATCH * NSEQ)      // 4096 token rows
#define ATT_SCALE 0.125f              // 1/sqrt(64)
#define LN_EPS   1e-5f

// ---------------------------------------------------------------- helpers
__device__ __forceinline__ v8f v8f_zero() {
  v8f z;
#pragma unroll
  for (int i = 0; i < 8; ++i) z[i] = 0.f;
  return z;
}

// DPP row_ror:N within each 16-lane row: lane i <- lane (i+N)%16 of its row.
// Pure VALU crossbar: no LDS round-trip, no dscnt waits.
template <int N>
__device__ __forceinline__ float dpp_ror(float v) {
  return __int_as_float(__builtin_amdgcn_update_dpp(
      0, __float_as_int(v), 0x120 | N, 0xf, 0xf, true));
}

// rotate-reduction: after +ror8,+ror4,+ror2,+ror1 every lane of a 16-lane
// row holds the row total (rotations compose to cover all 16 lanes).
__device__ __forceinline__ float half_sum16(float v) {
  v += dpp_ror<8>(v);
  v += dpp_ror<4>(v);
  v += dpp_ror<2>(v);
  v += dpp_ror<1>(v);
  return v;
}
__device__ __forceinline__ float half_max16(float v) {
  v = fmaxf(v, dpp_ror<8>(v));
  v = fmaxf(v, dpp_ror<4>(v));
  v = fmaxf(v, dpp_ror<2>(v));
  v = fmaxf(v, dpp_ror<1>(v));
  return v;
}
__device__ __forceinline__ float wave_sum32(float v) {
  v = half_sum16(v);
  return v + __shfl_xor(v, 16, 32);   // one cross-half hop only
}
__device__ __forceinline__ float wave_max32(float v) {
  v = half_max16(v);
  return fmaxf(v, __shfl_xor(v, 16, 32));
}

// A fragment (16x32 bf16, M x K): lanes 0-15 row M=lane, K {0..7,16..23};
// lanes 16-31 row M=lane-16, K {8..15,24..31}.  Row-major source, stride lda.
__device__ __forceinline__ v16bf load_a_frag(const __bf16* __restrict__ base,
                                             int lda, int m0, int k0, int lane) {
  const int half = lane >> 4, lr = lane & 15;
  const __bf16* p = base + (size_t)(m0 + lr) * lda + k0 + half * 8;
  v8bf lo = *(const v8bf*)p;
  v8bf hi = *(const v8bf*)(p + 16);
  v16bf r;
#pragma unroll
  for (int i = 0; i < 8; ++i) { r[i] = lo[i]; r[8 + i] = hi[i]; }
  return r;
}

// B fragment (32x16 bf16, K x N): lanes 0-15 col N=n0+lane, K {0..7,16..23};
// lanes 16-31 col N=n0+lane-16, K {8..15,24..31}.  Source holds B^T row-major
// (row n contiguous in k), stride ldb -> two 16B loads per lane.
__device__ __forceinline__ v16bf load_b_frag(const __bf16* __restrict__ base,
                                             int ldb, int n0, int k0, int lane) {
  const int half = lane >> 4, lr = lane & 15;
  const __bf16* p = base + (size_t)(n0 + lr) * ldb + k0 + half * 8;
  v8bf lo = *(const v8bf*)p;
  v8bf hi = *(const v8bf*)(p + 16);
  v16bf r;
#pragma unroll
  for (int i = 0; i < 8; ++i) { r[i] = lo[i]; r[8 + i] = hi[i]; }
  return r;
}

__device__ __forceinline__ v8f wmma_bf16(v16bf a, v16bf b, v8f c) {
  return __builtin_amdgcn_wmma_f32_16x16x32_bf16(false, a, false, b,
                                                 (short)0, c, false, false);
}

// ---------------------------------------------------------------- LN(in) -> bf16
__global__ void ln_in_kernel(const float* __restrict__ x,
                             const float* __restrict__ g,
                             const float* __restrict__ be,
                             __bf16* __restrict__ out) {
  const int row = blockIdx.x, lane = threadIdx.x;
  const float* xr = x + (size_t)row * DMODEL;
  float v[16], s = 0.f, s2 = 0.f;
#pragma unroll
  for (int i = 0; i < 16; ++i) {
    v[i] = xr[lane + i * 32];
    s += v[i]; s2 += v[i] * v[i];
  }
  s = wave_sum32(s); s2 = wave_sum32(s2);
  const float mu = s * (1.f / DMODEL);
  const float var = s2 * (1.f / DMODEL) - mu * mu;
  const float r = rsqrtf(var + LN_EPS);
#pragma unroll
  for (int i = 0; i < 16; ++i) {
    const int c = lane + i * 32;
    out[(size_t)row * DMODEL + c] = (__bf16)((v[i] - mu) * r * g[c] + be[c]);
  }
}

// ---------------------------------------------------------------- f32 -> bf16
__global__ void cvt_kernel(const float* __restrict__ src,
                           __bf16* __restrict__ dst, int n) {
  int i = blockIdx.x * blockDim.x + threadIdx.x;
  if (i < n) dst[i] = (__bf16)src[i];
}

// ---------------------------------------------------------------- WMMA GEMM
// y[M,N] = A[M,K=512] @ W[N,K=512]^T.  One wave computes a 32x64 tile:
// two A fragments share four B fragments (8 WMMAs / k-step), and the next
// k-step's fragments are loaded before the current WMMAs (double buffer) so
// the fully-unrolled loop overlaps global loads with the matrix pipe.
// MODE 0: f32 row-major out.  MODE 1: bf16 [B,H,N,64] (Q/K heads layout).
// MODE 2: bf16 [B,H,64,N]    (V transposed for the P@V B-fragments).
template <int MODE>
__global__ void gemm_wmma(const __bf16* __restrict__ A,
                          const __bf16* __restrict__ W,
                          void* __restrict__ Out) {
  constexpr int K = DMODEL;
  const int lane = threadIdx.x;
  const int m0 = blockIdx.x * 32;
  const int n0 = blockIdx.y * 64;
  const int half = lane >> 4, lr = lane & 15;

  v8f acc[2][4];
#pragma unroll
  for (int r = 0; r < 2; ++r)
#pragma unroll
    for (int t = 0; t < 4; ++t) acc[r][t] = v8f_zero();

  v16bf a0 = load_a_frag(A, K, m0,      0, lane);
  v16bf a1 = load_a_frag(A, K, m0 + 16, 0, lane);
  v16bf b0 = load_b_frag(W, K, n0,      0, lane);
  v16bf b1 = load_b_frag(W, K, n0 + 16, 0, lane);
  v16bf b2 = load_b_frag(W, K, n0 + 32, 0, lane);
  v16bf b3 = load_b_frag(W, K, n0 + 48, 0, lane);

#pragma unroll
  for (int k0 = 0; k0 < K; k0 += 32) {
    v16bf na0 = a0, na1 = a1, nb0 = b0, nb1 = b1, nb2 = b2, nb3 = b3;
    if (k0 + 32 < K) {  // issue next k-step's loads before this step's WMMAs
      na0 = load_a_frag(A, K, m0,      k0 + 32, lane);
      na1 = load_a_frag(A, K, m0 + 16, k0 + 32, lane);
      nb0 = load_b_frag(W, K, n0,      k0 + 32, lane);
      nb1 = load_b_frag(W, K, n0 + 16, k0 + 32, lane);
      nb2 = load_b_frag(W, K, n0 + 32, k0 + 32, lane);
      nb3 = load_b_frag(W, K, n0 + 48, k0 + 32, lane);
    }
    acc[0][0] = wmma_bf16(a0, b0, acc[0][0]);
    acc[0][1] = wmma_bf16(a0, b1, acc[0][1]);
    acc[0][2] = wmma_bf16(a0, b2, acc[0][2]);
    acc[0][3] = wmma_bf16(a0, b3, acc[0][3]);
    acc[1][0] = wmma_bf16(a1, b0, acc[1][0]);
    acc[1][1] = wmma_bf16(a1, b1, acc[1][1]);
    acc[1][2] = wmma_bf16(a1, b2, acc[1][2]);
    acc[1][3] = wmma_bf16(a1, b3, acc[1][3]);
    a0 = na0; a1 = na1; b0 = nb0; b1 = nb1; b2 = nb2; b3 = nb3;
  }

#pragma unroll
  for (int r = 0; r < 2; ++r) {
#pragma unroll
    for (int t = 0; t < 4; ++t) {
      const int n = n0 + t * 16 + lr;
#pragma unroll
      for (int j = 0; j < 8; ++j) {
        const int m = m0 + r * 16 + j + half * 8;  // C/D layout: VGPR j -> row j / j+8
        const float vv = acc[r][t][j];
        if constexpr (MODE == 0) {
          ((float*)Out)[(size_t)m * DMODEL + n] = vv;
        } else {
          const int b = m / NSEQ, tok = m % NSEQ, h = n >> 6, d = n & 63;
          if constexpr (MODE == 1)
            ((__bf16*)Out)[(((size_t)(b * NHEADS + h) * NSEQ) + tok) * HDIM + d] = (__bf16)vv;
          else
            ((__bf16*)Out)[(((size_t)(b * NHEADS + h) * HDIM) + d) * NSEQ + tok] = (__bf16)vv;
        }
      }
    }
  }
}

// ---------------------------------------------------------------- masks
__device__ __forceinline__ bool mask_ok(int q, int c, int kind, int w, int sh) {
  const int d = (q > c) ? (q - c) : (c - q);
  if (kind == 0) return true;                       // global
  if (kind == 1) return d <= w;                     // window
  return ((d & ((1 << sh) - 1)) == 0) && ((d >> sh) <= w);  // dilated
}

// ---------------------------------------------------------------- attention
// One wave per (b, h, 16-row q tile).  k-tiles of 32 -> P is one A fragment.
// V^T fragments load at the top of the tile so they arrive underneath the
// online-softmax work; softmax reductions are DPP-only (no LDS waits); exp
// values go straight to the LDS P-tile (no staging arrays -> no spills).
__global__ void attn_kernel(const __bf16* __restrict__ Qb,
                            const __bf16* __restrict__ Kb,
                            const __bf16* __restrict__ Vt,
                            __bf16* __restrict__ attn_out) {
  const int lane = threadIdx.x;
  const int qt = blockIdx.x, h = blockIdx.y, b = blockIdx.z;
  const int q0 = qt * 16;
  const size_t bh = (size_t)(b * NHEADS + h);
  const __bf16* Qh = Qb + bh * NSEQ * HDIM;   // [N,64] row-major
  const __bf16* Kh = Kb + bh * NSEQ * HDIM;   // [N,64] row-major
  const __bf16* Vh = Vt + bh * HDIM * NSEQ;   // [64,N] row-major (V^T)

  // head specs: 2 global | window 64/128/256 | dilated w=64 dil=2/4/8
  const int kinds[8] = {0, 0, 1, 1, 1, 2, 2, 2};
  const int wins[8]  = {0, 0, 64, 128, 256, 64, 64, 64};
  const int shs[8]   = {0, 0, 0, 0, 0, 1, 2, 3};
  const int kind = kinds[h], w = wins[h], sh = shs[h];
  const int reach = (kind == 0) ? NSEQ : ((kind == 1) ? w : (w << sh));
  int kmin = (kind == 0) ? 0 : max(0, q0 - reach);
  int kmax = (kind == 0) ? NSEQ : min(NSEQ, q0 + 16 + reach);
  kmin &= ~31;  // tile-skipping: only touch k-tiles inside the mask band

  const int half = lane >> 4, lr = lane & 15;
  const int myrow = half * 8;

  const v16bf qa0 = load_a_frag(Qh, HDIM, q0, 0, lane);
  const v16bf qa1 = load_a_frag(Qh, HDIM, q0, 32, lane);

  v8f o0 = v8f_zero(), o1 = v8f_zero(), o2 = v8f_zero(), o3 = v8f_zero();
  float mrow[8], lrow[8];
#pragma unroll
  for (int j = 0; j < 8; ++j) { mrow[j] = -1e30f; lrow[j] = 0.f; }

  __shared__ __bf16 Pl[16][32];

  for (int kt = kmin; kt < kmax; kt += 32) {
    // --- issue all of this tile's loads up front: 4 K frags then 4 V frags.
    // Loads retire in order, so the S WMMAs wait only on the K frags while
    // the V frags stream in during the softmax VALU work below.
    const v16bf kb0 = load_b_frag(Kh, HDIM, kt,      0,  lane);
    const v16bf kb1 = load_b_frag(Kh, HDIM, kt,      32, lane);
    const v16bf kb2 = load_b_frag(Kh, HDIM, kt + 16, 0,  lane);
    const v16bf kb3 = load_b_frag(Kh, HDIM, kt + 16, 32, lane);
    const v16bf vb0 = load_b_frag(Vh, NSEQ, 0,  kt, lane);
    const v16bf vb1 = load_b_frag(Vh, NSEQ, 16, kt, lane);
    const v16bf vb2 = load_b_frag(Vh, NSEQ, 32, kt, lane);
    const v16bf vb3 = load_b_frag(Vh, NSEQ, 48, kt, lane);

    // --- S = (Q K^T) over this 16x32 k-tile
    v8f s0 = v8f_zero(), s1 = v8f_zero();
    s0 = wmma_bf16(qa0, kb0, s0);
    s0 = wmma_bf16(qa1, kb1, s0);
    s1 = wmma_bf16(qa0, kb2, s1);
    s1 = wmma_bf16(qa1, kb3, s1);

    __syncthreads();  // previous iteration's P-tile reads are done

    // --- masked online softmax; rows split across wave halves per D layout.
    // All reductions are DPP rotate-reductions inside each 16-lane half.
    const int c0 = kt + lr, c1 = kt + 16 + lr;
#pragma unroll
    for (int j = 0; j < 8; ++j) {
      const int q = q0 + myrow + j;
      const bool ok0 = mask_ok(q, c0, kind, w, sh);
      const bool ok1 = mask_ok(q, c1, kind, w, sh);
      const float v0 = ok0 ? s0[j] * ATT_SCALE : -1e30f;
      const float v1 = ok1 ? s1[j] * ATT_SCALE : -1e30f;
      const float mx = half_max16(fmaxf(v0, v1));
      const float mnew = fmaxf(mrow[j], mx);
      const float alpha = __expf(mrow[j] - mnew);
      mrow[j] = mnew;
      const float e0 = ok0 ? __expf(v0 - mnew) : 0.f;
      const float e1 = ok1 ? __expf(v1 - mnew) : 0.f;
      lrow[j] = lrow[j] * alpha + half_sum16(e0 + e1);
      o0[j] *= alpha; o1[j] *= alpha; o2[j] *= alpha; o3[j] *= alpha;
      Pl[myrow + j][lr]      = (__bf16)e0;   // write P directly: no staging regs
      Pl[myrow + j][16 + lr] = (__bf16)e1;
    }
    __syncthreads();

    // --- read P back in A-fragment layout: two 16B LDS loads per lane
    const v8bf* prow = (const v8bf*)&Pl[lr][0];   // 4 x v8bf per row
    const v8bf plo = prow[half];                  // cols half*8 .. +7
    const v8bf phi = prow[2 + half];              // cols 16+half*8 .. +7
    v16bf pa;
#pragma unroll
    for (int i = 0; i < 8; ++i) { pa[i] = plo[i]; pa[8 + i] = phi[i]; }

    // --- O += P @ V  (V frags already resident)
    o0 = wmma_bf16(pa, vb0, o0);
    o1 = wmma_bf16(pa, vb1, o1);
    o2 = wmma_bf16(pa, vb2, o2);
    o3 = wmma_bf16(pa, vb3, o3);
  }

  // --- normalize and scatter to [B, N, H*64] bf16 for the Wo GEMM
#pragma unroll
  for (int j = 0; j < 8; ++j) {
    const float inv = 1.f / lrow[j];  // diagonal always unmasked -> lrow > 0
    const int tok = q0 + myrow + j;
    const size_t base = ((size_t)b * NSEQ + tok) * DMODEL + h * HDIM;
    attn_out[base + 0  + lr] = (__bf16)(o0[j] * inv);
    attn_out[base + 16 + lr] = (__bf16)(o1[j] * inv);
    attn_out[base + 32 + lr] = (__bf16)(o2[j] * inv);
    attn_out[base + 48 + lr] = (__bf16)(o3[j] * inv);
  }
}

// ---------------------------------------------------------------- gate logits
__global__ void gate_logit_kernel(const float* __restrict__ hpre,
                                  const float* __restrict__ b1,
                                  const float* __restrict__ w2,
                                  const float* __restrict__ b2,
                                  float* __restrict__ logits) {
  const int row = blockIdx.x, lane = threadIdx.x;
  float s = 0.f;
  for (int c = lane; c < DMODEL; c += 32)
    s += tanhf(hpre[(size_t)row * DMODEL + c] + b1[c]) * w2[c];
  s = wave_sum32(s);
  if (lane == 0) logits[row] = s + b2[0];
}

// ---------------------------------------------------------------- gate softmax
__global__ void tsoftmax_kernel(const float* __restrict__ logits,
                                float* __restrict__ T) {
  const int b = blockIdx.x, tid = threadIdx.x;
  const float* l = logits + (size_t)b * NSEQ;
  __shared__ float smax[8], ssum[8];
  float mx = -1e30f;
  for (int i = tid; i < NSEQ; i += 256) mx = fmaxf(mx, l[i]);
  mx = wave_max32(mx);
  if ((tid & 31) == 0) smax[tid >> 5] = mx;
  __syncthreads();
  mx = smax[0];
#pragma unroll
  for (int i = 1; i < 8; ++i) mx = fmaxf(mx, smax[i]);
  float s = 0.f;
  for (int i = tid; i < NSEQ; i += 256) s += __expf(l[i] - mx);
  s = wave_sum32(s);
  if ((tid & 31) == 0) ssum[tid >> 5] = s;
  __syncthreads();
  s = 0.f;
#pragma unroll
  for (int i = 0; i < 8; ++i) s += ssum[i];
  const float inv = 1.f / s;
  for (int i = tid; i < NSEQ; i += 256)
    T[(size_t)b * NSEQ + i] = __expf(l[i] - mx) * inv;
}

// ---------------------------------------------------------------- gate*out + residual + LN(out)
__global__ void final_kernel(const float* __restrict__ aout,
                             const float* __restrict__ tg,
                             const float* __restrict__ x0,
                             const float* __restrict__ g,
                             const float* __restrict__ be,
                             float* __restrict__ out) {
  const int row = blockIdx.x, lane = threadIdx.x;
  const float t = tg[row];
  float v[16], s = 0.f, s2 = 0.f;
#pragma unroll
  for (int i = 0; i < 16; ++i) {
    const int c = lane + i * 32;
    v[i] = aout[(size_t)row * DMODEL + c] * t + x0[(size_t)row * DMODEL + c];
    s += v[i]; s2 += v[i] * v[i];
  }
  s = wave_sum32(s); s2 = wave_sum32(s2);
  const float mu = s * (1.f / DMODEL);
  const float var = s2 * (1.f / DMODEL) - mu * mu;
  const float r = rsqrtf(var + LN_EPS);
#pragma unroll
  for (int i = 0; i < 16; ++i) {
    const int c = lane + i * 32;
    out[(size_t)row * DMODEL + c] = (v[i] - mu) * r * g[c] + be[c];
  }
}

// ---------------------------------------------------------------- launcher
extern "C" void kernel_launch(void* const* d_in, const int* in_sizes, int n_in,
                              void* d_out, int out_size, void* d_ws, size_t ws_size,
                              hipStream_t stream) {
  (void)in_sizes; (void)n_in; (void)out_size; (void)ws_size;
  const float* x     = (const float*)d_in[0];
  const float* Wq    = (const float*)d_in[1];
  const float* Wk    = (const float*)d_in[2];
  const float* Wv    = (const float*)d_in[3];
  const float* Wo    = (const float*)d_in[4];
  const float* T_w1  = (const float*)d_in[5];
  const float* T_b1  = (const float*)d_in[6];
  const float* T_w2  = (const float*)d_in[7];
  const float* T_b2  = (const float*)d_in[8];
  const float* lng_i = (const float*)d_in[9];
  const float* lnb_i = (const float*)d_in[10];
  const float* lng_o = (const float*)d_in[11];
  const float* lnb_o = (const float*)d_in[12];

  char* ws = (char*)d_ws;
  size_t off = 0;
  auto alloc = [&](size_t bytes) -> char* {
    char* p = ws + off;
    off += (bytes + 255) & ~(size_t)255;
    return p;
  };
  const size_t WMAT = (size_t)DMODEL * DMODEL;                      // 262144
  const size_t HEADELEMS = (size_t)NBATCH * NHEADS * NSEQ * HDIM;   // 2M

  __bf16* xnb   = (__bf16*)alloc((size_t)NROWS * DMODEL * 2);
  __bf16* wqb   = (__bf16*)alloc(WMAT * 2);
  __bf16* wkb   = (__bf16*)alloc(WMAT * 2);
  __bf16* wvb   = (__bf16*)alloc(WMAT * 2);
  __bf16* wob   = (__bf16*)alloc(WMAT * 2);
  __bf16* w1b   = (__bf16*)alloc(WMAT * 2);
  __bf16* qb    = (__bf16*)alloc(HEADELEMS * 2);
  __bf16* kb    = (__bf16*)alloc(HEADELEMS * 2);
  __bf16* vt    = (__bf16*)alloc(HEADELEMS * 2);
  __bf16* attnb = (__bf16*)alloc((size_t)NROWS * DMODEL * 2);
  float*  aoutf = (float*)alloc((size_t)NROWS * DMODEL * 4);
  float*  logit = (float*)alloc((size_t)NROWS * 4);
  float*  tgate = (float*)alloc((size_t)NROWS * 4);
  float*  hpre  = (float*)qb;  // 8 MB alias over qb+kb (dead after attention)

  // 1) input layernorm -> bf16
  ln_in_kernel<<<NROWS, 32, 0, stream>>>(x, lng_i, lnb_i, xnb);

  // 2) weights -> bf16
  cvt_kernel<<<(int)((WMAT + 255) / 256), 256, 0, stream>>>(Wq, wqb, (int)WMAT);
  cvt_kernel<<<(int)((WMAT + 255) / 256), 256, 0, stream>>>(Wk, wkb, (int)WMAT);
  cvt_kernel<<<(int)((WMAT + 255) / 256), 256, 0, stream>>>(Wv, wvb, (int)WMAT);
  cvt_kernel<<<(int)((WMAT + 255) / 256), 256, 0, stream>>>(Wo, wob, (int)WMAT);
  cvt_kernel<<<(int)((WMAT + 255) / 256), 256, 0, stream>>>(T_w1, w1b, (int)WMAT);

  // 3) QKV projections (WMMA): Q/K -> [B,H,N,64], V -> [B,H,64,N]
  dim3 ggrid(NROWS / 32, DMODEL / 64);
  gemm_wmma<1><<<ggrid, 32, 0, stream>>>(xnb, wqb, qb);
  gemm_wmma<1><<<ggrid, 32, 0, stream>>>(xnb, wkb, kb);
  gemm_wmma<2><<<ggrid, 32, 0, stream>>>(xnb, wvb, vt);

  // 4) masked flash attention (tile-skipping per head spec)
  attn_kernel<<<dim3(NSEQ / 16, NHEADS, NBATCH), 32, 0, stream>>>(qb, kb, vt, attnb);

  // 5) output projection (WMMA, f32 out)
  gemm_wmma<0><<<ggrid, 32, 0, stream>>>(attnb, wob, aoutf);

  // 6) gate path: hidden = xn @ T_w1^T (WMMA, reuses qb/kb scratch)
  gemm_wmma<0><<<ggrid, 32, 0, stream>>>(xnb, w1b, hpre);
  gate_logit_kernel<<<NROWS, 32, 0, stream>>>(hpre, T_b1, T_w2, T_b2, logit);
  tsoftmax_kernel<<<NBATCH, 256, 0, stream>>>(logit, tgate);

  // 7) gate * out + residual + output layernorm
  final_kernel<<<NROWS, 32, 0, stream>>>(aoutf, tgate, x, lng_o, lnb_o, (float*)d_out);
}
